// NnstGenerate_16329465659491
// MI455X (gfx1250) — matrix operand order repaired and checked by reading, not compiled, not run on hardware
//
#include <hip/hip_runtime.h>
#include <hip/hip_bf16.h>

// Problem constants (from reference)
#define N_C  16384
#define N_S  16384
#define DIM  2688
// GEMM tiling
#define BM   128
#define BN   128
#define BK   32
#define LDSW 40   // padded LDS row stride (bf16 elems) -> conflict-free b128 frag loads

typedef __attribute__((ext_vector_type(16))) __bf16 v16bf;
typedef __attribute__((ext_vector_type(8)))  __bf16 v8bf;
typedef __attribute__((ext_vector_type(4)))  __bf16 v4bf;
typedef __attribute__((ext_vector_type(8)))  float  v8f;

union FragBf { v16bf v; v8bf h[2]; };

// ---- gfx1250 async global->LDS path (guarded: falls back to reg staging) ----
#if defined(__AMDGCN__) && __has_builtin(__builtin_amdgcn_global_load_async_to_lds_b128)
#define USE_ASYNC_LDS 1
typedef int v4i_vs __attribute__((__vector_size__(16)));
typedef __attribute__((address_space(1))) void*    gas_void;
typedef __attribute__((address_space(3))) void*    las_void;
typedef __attribute__((address_space(1))) v4i_vs*  gvec_ptr;
typedef __attribute__((address_space(3))) v4i_vs*  lvec_ptr;

__device__ __forceinline__ gvec_ptr as_global_v4i(const void* p) {
    return (gvec_ptr)(gas_void)(void*)p;
}
__device__ __forceinline__ lvec_ptr as_lds_v4i(void* p) {
    return (lvec_ptr)(las_void)p;
}
__device__ __forceinline__ void async_b128(const void* g, void* l) {
    __builtin_amdgcn_global_load_async_to_lds_b128(as_global_v4i(g), as_lds_v4i(l), 0, 0);
}
#else
#define USE_ASYNC_LDS 0
#endif

__device__ __forceinline__ void wait_async_all() {
#if defined(__AMDGCN__)
#if __has_builtin(__builtin_amdgcn_s_wait_asynccnt)
    __builtin_amdgcn_s_wait_asynccnt(0);
#elif USE_ASYNC_LDS
    asm volatile("s_wait_asynccnt 0" ::: "memory");
#endif
#endif
}

// Monotone float->uint key: larger float => larger key (cos sims are finite)
__device__ __forceinline__ unsigned long long packKey(float f, int sidx) {
    unsigned int u = __float_as_uint(f);
    unsigned int k = (u & 0x80000000u) ? ~u : (u | 0x80000000u);
    // ~idx in low word: on equal value, larger packed => smaller index (jnp.argmax tie rule)
    return ((unsigned long long)k << 32) | (unsigned int)(~(unsigned int)sidx);
}

__device__ __forceinline__ unsigned long long shflxor64(unsigned long long v, int m) {
    unsigned int lo = __shfl_xor((unsigned int)v, m, 32);
    unsigned int hi = __shfl_xor((unsigned int)(v >> 32), m, 32);
    return ((unsigned long long)hi << 32) | lo;
}

// ---------------------------------------------------------------------------
// Kernel 1: row-normalize (x * rsqrt(sum(x^2)+eps)) and convert to bf16.
// grid.x = N_C + N_S ; rows < N_C come from feats_c, rest from feats_s.
// ---------------------------------------------------------------------------
__global__ __launch_bounds__(256) void nnst_normalize(
    const float* __restrict__ fc, const float* __restrict__ fs,
    __bf16* __restrict__ cn, __bf16* __restrict__ sn)
{
    const int row = blockIdx.x;
    const int tid = threadIdx.x;
    const bool isC = row < N_C;
    const float* src = isC ? (fc + (size_t)row * DIM)
                           : (fs + (size_t)(row - N_C) * DIM);
    __bf16* dst = isC ? (cn + (size_t)row * DIM)
                      : (sn + (size_t)(row - N_C) * DIM);

    const float4* s4 = (const float4*)src;
    float ssq = 0.f;
    for (int j = tid; j < DIM / 4; j += 256) {
        float4 v = s4[j];
        ssq += v.x * v.x + v.y * v.y + v.z * v.z + v.w * v.w;
    }
    #pragma unroll
    for (int off = 16; off; off >>= 1) ssq += __shfl_xor(ssq, off, 32);

    __shared__ float red[8];
    if ((tid & 31) == 0) red[tid >> 5] = ssq;
    __syncthreads();
    float tot = 0.f;
    #pragma unroll
    for (int i = 0; i < 8; ++i) tot += red[i];
    const float sc = rsqrtf(tot + 1e-8f);

    v4bf* d4 = (v4bf*)dst;
    for (int j = tid; j < DIM / 4; j += 256) {
        float4 v = s4[j];
        v4bf o;
        o[0] = (__bf16)(v.x * sc); o[1] = (__bf16)(v.y * sc);
        o[2] = (__bf16)(v.z * sc); o[3] = (__bf16)(v.w * sc);
        d4[j] = o;
    }
}

// ---------------------------------------------------------------------------
// Kernel 2: zero the packed (key|~idx) argmax accumulator (identity = 0).
// ---------------------------------------------------------------------------
__global__ void nnst_init_best(unsigned long long* __restrict__ best) {
    int i = blockIdx.x * blockDim.x + threadIdx.x;
    if (i < N_C) best[i] = 0ull;
}

// ---------------------------------------------------------------------------
// Kernel 3: fused 128x128-tile bf16 WMMA GEMM (sim = cn . sn^T) + row argmax.
// 8 waves: 4 along M x 2 along N; each wave: 2x4 v_wmma_f32_16x16x32_bf16.
// Double-buffered LDS; next tile staged via async global->LDS DMA (ASYNCcnt)
// when available, else via VGPR staging. Argmax fused via u64 atomicMax.
// ---------------------------------------------------------------------------
__global__ __launch_bounds__(256) void nnst_sim_argmax(
    const __bf16* __restrict__ cn, const __bf16* __restrict__ sn,
    unsigned long long* __restrict__ best)
{
    __shared__ __bf16 As[2][BM * LDSW];   // 2 * 128*40*2B = 20 KB
    __shared__ __bf16 Bs[2][BN * LDSW];   // 20 KB

    const int tid   = threadIdx.x;
    const int lane  = tid & 31;
    const int l15   = lane & 15;
    const int hi    = lane >> 4;          // half-wave select
    const int wave  = tid >> 5;
    const int waveM = wave & 3;           // 0..3  -> 32-row group
    const int waveN = wave >> 2;          // 0..1  -> 64-col group
    const int mBlock = blockIdx.y * BM;
    const int nBlock = blockIdx.x * BN;

    // staging: thread moves 16 bf16 (32B) per matrix per K-tile
    const int ldRow = tid >> 1;
    const int ldCol = (tid & 1) * 16;
    const int lofs  = ldRow * LDSW + ldCol;
    const __bf16* ag = cn + (size_t)(mBlock + ldRow) * DIM + ldCol;
    const __bf16* bg = sn + (size_t)(nBlock + ldRow) * DIM + ldCol;

#if USE_ASYNC_LDS
    // Async DMA staging: 4x GLOBAL_LOAD_ASYNC_TO_LDS_B128 per thread per tile.
    {
        __bf16* dA = &As[0][lofs];
        __bf16* dB = &Bs[0][lofs];
        async_b128(ag,     dA);
        async_b128(ag + 8, dA + 8);
        async_b128(bg,     dB);
        async_b128(bg + 8, dB + 8);
    }
    wait_async_all();
    __syncthreads();
#else
    {
        int4 ra0 = *(const int4*)(ag);     int4 ra1 = *(const int4*)(ag + 8);
        int4 rb0 = *(const int4*)(bg);     int4 rb1 = *(const int4*)(bg + 8);
        *(int4*)(&As[0][lofs]) = ra0;      *(int4*)(&As[0][lofs + 8]) = ra1;
        *(int4*)(&Bs[0][lofs]) = rb0;      *(int4*)(&Bs[0][lofs + 8]) = rb1;
    }
    __syncthreads();
#endif

    v8f zero = 0.f;
    v8f acc[2][4];
    #pragma unroll
    for (int i = 0; i < 2; ++i)
        #pragma unroll
        for (int j = 0; j < 4; ++j) acc[i][j] = zero;

    int buf = 0;
    for (int k0 = 0; k0 < DIM; k0 += BK) {
        const bool pre = (k0 + BK) < DIM;

#if USE_ASYNC_LDS
        if (pre) {  // DMA next tile into buf^1; runs in the shadow of the WMMAs
            const __bf16* an = ag + (k0 + BK);
            const __bf16* bn = bg + (k0 + BK);
            __bf16* dA = &As[buf ^ 1][lofs];
            __bf16* dB = &Bs[buf ^ 1][lofs];
            async_b128(an,     dA);
            async_b128(an + 8, dA + 8);
            async_b128(bn,     dB);
            async_b128(bn + 8, dB + 8);
        }
#else
        int4 na0{}, na1{}, nb0{}, nb1{};
        if (pre) {  // issue next-tile global loads early (hide latency)
            const __bf16* an = ag + (k0 + BK);
            const __bf16* bn = bg + (k0 + BK);
            na0 = *(const int4*)(an);  na1 = *(const int4*)(an + 8);
            nb0 = *(const int4*)(bn);  nb1 = *(const int4*)(bn + 8);
        }
#endif

        // A frags: lane(0-15) K={0-7,16-23}; lane(16-31) K={8-15,24-31}
        FragBf a[2], b[4];
        #pragma unroll
        for (int mf = 0; mf < 2; ++mf) {
            const __bf16* p = &As[buf][(waveM * 32 + mf * 16 + l15) * LDSW + hi * 8];
            a[mf].h[0] = *(const v8bf*)(p);
            a[mf].h[1] = *(const v8bf*)(p + 16);
        }
        // B frags: lane n = column, lanes 0-15 K={0-7,16-23}, lanes 16-31 K={8-15,24-31}
        #pragma unroll
        for (int nf = 0; nf < 4; ++nf) {
            const __bf16* p = &Bs[buf][(waveN * 64 + nf * 16 + l15) * LDSW + hi * 8];
            b[nf].h[0] = *(const v8bf*)(p);
            b[nf].h[1] = *(const v8bf*)(p + 16);
        }

        #pragma unroll
        for (int mf = 0; mf < 2; ++mf)
            #pragma unroll
            for (int nf = 0; nf < 4; ++nf)
                acc[mf][nf] = __builtin_amdgcn_wmma_f32_16x16x32_bf16(
                    false, a[mf].v, false, b[nf].v,
                    (short)0, acc[mf][nf], false, false);

        if (pre) {
#if USE_ASYNC_LDS
            wait_async_all();       // own DMA done
            __syncthreads();        // everyone's DMA done
#else
            __bf16* dA = &As[buf ^ 1][lofs];
            __bf16* dB = &Bs[buf ^ 1][lofs];
            *(int4*)dA = na0;  *(int4*)(dA + 8) = na1;
            *(int4*)dB = nb0;  *(int4*)(dB + 8) = nb1;
            __syncthreads();
#endif
            buf ^= 1;
        }
    }

    // Epilogue: per content row, max over this wave's 64 style columns,
    // then combine globally with one u64 atomicMax per (row, half-wave).
    // C/D layout: VGPR r, lanes 0-15 -> M = base+r, lanes 16-31 -> M = base+8+r; N = l15.
    #pragma unroll
    for (int mf = 0; mf < 2; ++mf) {
        #pragma unroll
        for (int r = 0; r < 8; ++r) {
            unsigned long long pk = 0ull;
            #pragma unroll
            for (int nf = 0; nf < 4; ++nf) {
                float f = acc[mf][nf][r];
                int n = nBlock + waveN * 64 + nf * 16 + l15;
                unsigned long long q = packKey(f, n);
                pk = pk > q ? pk : q;
            }
            #pragma unroll
            for (int off = 8; off; off >>= 1) {
                unsigned long long q = shflxor64(pk, off);
                pk = pk > q ? pk : q;
            }
            if (l15 == 0) {
                int m = mBlock + waveM * 32 + mf * 16 + hi * 8 + r;
                atomicMax(&best[m], pk);
            }
        }
    }
}

// ---------------------------------------------------------------------------
// Kernel 4: gather un-normalized style rows by argmax index.
// ---------------------------------------------------------------------------
__global__ __launch_bounds__(256) void nnst_gather(
    const float* __restrict__ fs, const unsigned long long* __restrict__ best,
    float* __restrict__ out)
{
    const int c = blockIdx.x;
    const unsigned int sidx = ~(unsigned int)(best[c] & 0xffffffffull);
    const float4* s = (const float4*)(fs + (size_t)sidx * DIM);
    float4*       o = (float4*)(out + (size_t)c * DIM);
    for (int j = threadIdx.x; j < DIM / 4; j += 256) o[j] = s[j];
}

// ---------------------------------------------------------------------------
// Workspace layout (requires ws_size >= 176,291,840 bytes):
//   [0)                      cn bf16  : N_C*DIM*2 = 88,080,384
//   [88,080,384)             sn bf16  : N_S*DIM*2 = 88,080,384
//   [176,160,768)            best u64 : N_C*8     =    131,072
// ---------------------------------------------------------------------------
extern "C" void kernel_launch(void* const* d_in, const int* in_sizes, int n_in,
                              void* d_out, int out_size, void* d_ws, size_t ws_size,
                              hipStream_t stream)
{
    const float* fc = (const float*)d_in[0];
    const float* fs = (const float*)d_in[1];
    float* out = (float*)d_out;

    char* ws = (char*)d_ws;
    __bf16* cn = (__bf16*)ws;
    __bf16* sn = (__bf16*)(ws + (size_t)N_C * DIM * 2);
    unsigned long long* best =
        (unsigned long long*)(ws + (size_t)(N_C + N_S) * DIM * 2);

    nnst_normalize<<<N_C + N_S, 256, 0, stream>>>(fc, fs, cn, sn);
    nnst_init_best<<<(N_C + 255) / 256, 256, 0, stream>>>(best);
    dim3 grid(N_S / BN, N_C / BM);
    nnst_sim_argmax<<<grid, 256, 0, stream>>>(cn, sn, best);
    nnst_gather<<<N_C, 256, 0, stream>>>(fs, best, out);
}
